// Edge_Readout_only_atom_embedding_87634512707842
// MI455X (gfx1250) — matrix operand order repaired and checked by reading, not compiled, not run on hardware
//
#include <hip/hip_runtime.h>

// ---------------- problem constants (match reference) ----------------
#define N_ATOMS 200000
#define NBRS    10
#define HDIM    128
#define FDIM    151
#define FEAT    200
#define FFN_H   512
#define TASKS   12
#define EPSLN   1e-6f

#define K1PAD   288   // 279 -> 288 (9 x 32)
#define K2PAD   512
#define KFPAD   352   // 328 -> 352 (11 x 32)
#define NFPAD   16    // 12 -> 16

typedef __attribute__((ext_vector_type(16))) __bf16 v16bf;
typedef __attribute__((ext_vector_type(8)))  float  v8f;

union Frag { uint4 q[2]; v16bf v; };

__device__ __forceinline__ unsigned short f2bf(float f) {
  unsigned int u = __builtin_bit_cast(unsigned int, f);
  u += 0x7FFFu + ((u >> 16) & 1u);          // round to nearest even
  return (unsigned short)(u >> 16);
}

// ---------------- weight conversion: row-major f32 [K][N] -> bf16 col-major [Npad][Kpad] ----
__global__ void convert_wcm_kernel(const float* __restrict__ W, unsigned short* __restrict__ out,
                                   int K, int N, int Kpad, int Npad) {
  size_t i = (size_t)blockIdx.x * blockDim.x + threadIdx.x;
  size_t total = (size_t)Npad * Kpad;
  if (i >= total) return;
  int n = (int)(i / Kpad), k = (int)(i % Kpad);
  out[i] = (k < K && n < N) ? f2bf(W[(size_t)k * N + n]) : (unsigned short)0;
}

// ---------------- neighbor gather-sum + concat + bf16 pack: one wave per atom ----------------
__global__ void prep_atoms_kernel(const float* __restrict__ hidden,
                                  const float* __restrict__ orig,
                                  const int*   __restrict__ nbr,
                                  unsigned short* __restrict__ Xout) {
  int atom = blockIdx.x * (blockDim.x >> 5) + ((int)threadIdx.x >> 5);
  int lane = (int)threadIdx.x & 31;
  if (atom >= N_ATOMS) return;
  const int* nb = nbr + (size_t)atom * NBRS;
  float s0 = 0.f, s1 = 0.f, s2 = 0.f, s3 = 0.f;
  for (int j = 0; j < NBRS; ++j) {
    const float* hr = hidden + (size_t)nb[j] * HDIM;
    s0 += hr[lane]; s1 += hr[lane + 32]; s2 += hr[lane + 64]; s3 += hr[lane + 96];
  }
  unsigned short* xr = Xout + (size_t)atom * K1PAD;
  const float* orow = orig + (size_t)atom * FDIM;
  for (int c = lane; c < FDIM; c += 32) xr[c] = f2bf(orow[c]);
  xr[FDIM + lane]      = f2bf(s0);
  xr[FDIM + lane + 32] = f2bf(s1);
  xr[FDIM + lane + 64] = f2bf(s2);
  xr[FDIM + lane + 96] = f2bf(s3);
  if (lane < K1PAD - (FDIM + HDIM)) xr[FDIM + HDIM + lane] = 0;
}

// ---------------- generic WMMA bf16 GEMM: one wave per 16 x (16*NT) output strip --------------
// X   : [M][KP] bf16 row-major
// Wcm : [Npad][KP] bf16 column-major
// NT  : 16-col tiles per wave (A fragment reused NT times per K-step)
// KP  : compile-time padded K -> fully unrolled, double-buffered K pipeline
// epi : 0 = +bias,ReLU -> bf16 out [M][Npad]
//       1 = +bias      -> f32  out [M][Npad]
//       2 = +bias,sigmoid -> f32 out [M][nValid] (columns >= nValid dropped)
template <int NT, int KP>
__global__ void wmma_gemm_kernel(const unsigned short* __restrict__ X,
                                 const unsigned short* __restrict__ Wcm,
                                 const float* __restrict__ bias,
                                 void* __restrict__ out,
                                 int M, int Npad, int nValid, int epi) {
  int wave = (int)((blockIdx.x * blockDim.x + threadIdx.x) >> 5);
  int lane = (int)threadIdx.x & 31;
  int ngroups = (Npad >> 4) / NT;            // groups of NT tiles along N
  int total   = (M >> 4) * ngroups;
  if (wave >= total) return;                 // whole-wave uniform exit: EXEC all-1s for WMMA
  int mt = wave / ngroups, ng = wave % ngroups;
  int khalf = lane >> 4;
  int row     = (mt << 4) + (lane & 15);     // A: lane holds row (lane&15), K-half by lane>>4
  int colbase = ng * (NT << 4) + (lane & 15);// B: lane holds column, 16 consecutive K per half
  const unsigned short* xrow = X + (size_t)row * KP + khalf * 16;
  const unsigned short* wcol[NT];
  #pragma unroll
  for (int i = 0; i < NT; ++i)
    wcol[i] = Wcm + (size_t)(colbase + (i << 4)) * KP + khalf * 16;

  v8f acc[NT];
  #pragma unroll
  for (int i = 0; i < NT; ++i) acc[i] = (v8f){};

  auto loadk = [&](int k0, Frag& a, Frag (&b)[NT]) {
    a.q[0] = *(const uint4*)(xrow + k0);        // K = k0 + 16*khalf + [0..7]
    a.q[1] = *(const uint4*)(xrow + k0 + 8);    // K = k0 + 16*khalf + 8 + [0..7]
    #pragma unroll
    for (int i = 0; i < NT; ++i) {
      const uint4* wp = (const uint4*)(wcol[i] + k0);
      b[i].q[0] = wp[0]; b[i].q[1] = wp[1];     // K = k0 + 16*khalf + [0..15]
    }
  };
  auto wmmas = [&](Frag& a, Frag (&b)[NT]) {
    #pragma unroll
    for (int i = 0; i < NT; ++i)
      acc[i] = __builtin_amdgcn_wmma_f32_16x16x32_bf16(false, a.v, false, b[i].v,
                                                       (short)0, acc[i], false, false);
  };

  constexpr int NK = KP / 32;
  Frag a0, a1; Frag b0[NT], b1[NT];
  loadk(0, a0, b0);                // prime buffer 0
  #pragma unroll
  for (int s = 0; s < NK; ++s) {   // compile-time unrolled double-buffer pipeline
    if (s + 1 < NK) {
      if (s & 1) loadk((s + 1) * 32, a0, b0);
      else       loadk((s + 1) * 32, a1, b1);
    }
    if (s & 1) wmmas(a1, b1);
    else       wmmas(a0, b0);
  }

  #pragma unroll
  for (int i = 0; i < NT; ++i) {
    int n = colbase + (i << 4);
    float bn = (n < nValid) ? bias[n] : 0.f;
    #pragma unroll
    for (int r = 0; r < 8; ++r) {
      int m = (mt << 4) + (khalf << 3) + r;     // C/D: VGPR r -> row r (+8 for hi lanes)
      float v = acc[i][r] + bn;
      if (epi == 0) {
        v = v > 0.f ? v : 0.f;
        ((unsigned short*)out)[(size_t)m * Npad + n] = f2bf(v);
      } else if (epi == 1) {
        ((float*)out)[(size_t)m * Npad + n] = v;
      } else {
        if (n < nValid)
          ((float*)out)[(size_t)m * nValid + n] = 1.f / (1.f + __expf(-v));
      }
    }
  }
}

// ---------------- wave-per-row LayerNorm over 128 columns ----------------
__global__ void layernorm128_kernel(const float* __restrict__ in,
                                    const float* __restrict__ gamma,
                                    const float* __restrict__ beta,
                                    float* __restrict__ out, int M) {
  int row = blockIdx.x * (blockDim.x >> 5) + ((int)threadIdx.x >> 5);
  int lane = (int)threadIdx.x & 31;
  if (row >= M) return;
  const float* r = in + (size_t)row * HDIM;
  float v0 = r[lane], v1 = r[lane + 32], v2 = r[lane + 64], v3 = r[lane + 96];
  float s = v0 + v1 + v2 + v3;
  for (int o = 16; o > 0; o >>= 1) s += __shfl_xor(s, o, 32);
  float mean = s * (1.f / 128.f);
  float d0 = v0 - mean, d1 = v1 - mean, d2 = v2 - mean, d3 = v3 - mean;
  float q = d0 * d0 + d1 * d1 + d2 * d2 + d3 * d3;
  for (int o = 16; o > 0; o >>= 1) q += __shfl_xor(q, o, 32);
  float rstd = __frsqrt_rn(q * (1.f / 128.f) + EPSLN);
  float* w = out + (size_t)row * HDIM;
  w[lane]      = d0 * rstd * gamma[lane]      + beta[lane];
  w[lane + 32] = d1 * rstd * gamma[lane + 32] + beta[lane + 32];
  w[lane + 64] = d2 * rstd * gamma[lane + 64] + beta[lane + 64];
  w[lane + 96] = d3 * rstd * gamma[lane + 96] + beta[lane + 96];
}

// ---------------- misc: zero, segment atomic sum, mol concat ----------------
__global__ void zero_kernel(float* p, size_t n) {
  size_t i = (size_t)blockIdx.x * blockDim.x + threadIdx.x;
  if (i < n) p[i] = 0.f;
}

__global__ void seg_reduce_kernel(const float* __restrict__ x, const int* __restrict__ seg,
                                  float* __restrict__ mol_sum, float* __restrict__ cnt) {
  int atom = blockIdx.x * (blockDim.x >> 5) + ((int)threadIdx.x >> 5);
  int lane = (int)threadIdx.x & 31;
  if (atom >= N_ATOMS) return;
  int s = seg[atom];
  const float* r = x + (size_t)atom * HDIM;
  float* m = mol_sum + (size_t)s * HDIM;
  atomicAdd(m + lane,      r[lane]);
  atomicAdd(m + lane + 32, r[lane + 32]);
  atomicAdd(m + lane + 64, r[lane + 64]);
  atomicAdd(m + lane + 96, r[lane + 96]);
  if (lane == 0) atomicAdd(cnt + s, 1.0f);
}

__global__ void mol_prep_kernel(const float* __restrict__ mol_sum, const float* __restrict__ cnt,
                                const float* __restrict__ feat, unsigned short* __restrict__ molX,
                                int nMols) {
  int mol = blockIdx.x * (blockDim.x >> 5) + ((int)threadIdx.x >> 5);
  int lane = (int)threadIdx.x & 31;
  if (mol >= nMols) return;
  float c = cnt[mol];
  float inv = 1.f / (c > 1.f ? c : 1.f);
  unsigned short* xr = molX + (size_t)mol * KFPAD;
  const float* ms = mol_sum + (size_t)mol * HDIM;
  for (int i = lane; i < HDIM; i += 32) xr[i] = f2bf(ms[i] * inv);
  const float* f = feat + (size_t)mol * FEAT;
  for (int i = lane; i < FEAT; i += 32) xr[HDIM + i] = f2bf(f[i]);
  if (lane < KFPAD - (HDIM + FEAT)) xr[HDIM + FEAT + lane] = 0;
}

// ---------------- host side ----------------
static inline unsigned int cdiv(size_t a, unsigned int b) { return (unsigned int)((a + b - 1) / b); }

extern "C" void kernel_launch(void* const* d_in, const int* in_sizes, int n_in,
                              void* d_out, int out_size, void* d_ws, size_t ws_size,
                              hipStream_t stream) {
  const float* atom_output = (const float*)d_in[0];
  const float* bond_output = (const float*)d_in[1];
  const float* orig        = (const float*)d_in[2];
  const float* feats       = (const float*)d_in[3];
  const int*   a2a         = (const int*)d_in[4];
  const int*   a2b         = (const int*)d_in[5];
  const int*   seg         = (const int*)d_in[6];
  const float* W1_aa = (const float*)d_in[7];
  const float* b1_aa = (const float*)d_in[8];
  const float* W2_aa = (const float*)d_in[9];
  const float* b2_aa = (const float*)d_in[10];
  const float* W1_ab = (const float*)d_in[11];
  const float* b1_ab = (const float*)d_in[12];
  const float* W2_ab = (const float*)d_in[13];
  const float* b2_ab = (const float*)d_in[14];
  const float* gamma_a = (const float*)d_in[15];
  const float* beta_a  = (const float*)d_in[16];
  const float* gamma_b = (const float*)d_in[17];
  const float* beta_b  = (const float*)d_in[18];
  const float* Wf1 = (const float*)d_in[19];
  const float* bf1 = (const float*)d_in[20];
  const float* Wf2 = (const float*)d_in[21];
  const float* bf2 = (const float*)d_in[22];

  float* out_mol  = (float*)d_out;                         // [10000][12]
  float* out_atom = (float*)d_out + (size_t)10000 * TASKS; // [200000][128]

  // ---- workspace layout (bytes, 256-aligned) ----
  char* ws = (char*)d_ws;
  size_t off = 0;
  auto alloc = [&](size_t bytes) { size_t o = off; off += (bytes + 255) & ~(size_t)255; return o; };
  size_t o_w1aa = alloc((size_t)FFN_H * K1PAD * 2);
  size_t o_w2aa = alloc((size_t)HDIM  * K2PAD * 2);
  size_t o_w1ab = alloc((size_t)FFN_H * K1PAD * 2);
  size_t o_w2ab = alloc((size_t)HDIM  * K2PAD * 2);
  size_t o_wf1  = alloc((size_t)FFN_H * KFPAD * 2);
  size_t o_wf2  = alloc((size_t)NFPAD * K2PAD * 2);
  size_t o_X    = alloc((size_t)N_ATOMS * K1PAD * 2);
  size_t o_H1   = alloc((size_t)N_ATOMS * FFN_H * 2);
  size_t o_F2   = alloc((size_t)N_ATOMS * HDIM * 4);
  size_t o_msum = alloc((size_t)10000 * HDIM * 4);
  size_t o_cnt  = alloc((size_t)10000 * 4);
  size_t o_molX = alloc((size_t)10000 * KFPAD * 2);
  size_t o_MH   = alloc((size_t)10000 * FFN_H * 2);
  (void)ws_size; (void)in_sizes; (void)n_in; (void)out_size;

  unsigned short* w1aa = (unsigned short*)(ws + o_w1aa);
  unsigned short* w2aa = (unsigned short*)(ws + o_w2aa);
  unsigned short* w1ab = (unsigned short*)(ws + o_w1ab);
  unsigned short* w2ab = (unsigned short*)(ws + o_w2ab);
  unsigned short* wf1  = (unsigned short*)(ws + o_wf1);
  unsigned short* wf2  = (unsigned short*)(ws + o_wf2);
  unsigned short* Xbuf = (unsigned short*)(ws + o_X);
  unsigned short* H1   = (unsigned short*)(ws + o_H1);
  float*          F2   = (float*)(ws + o_F2);
  float*          msum = (float*)(ws + o_msum);
  float*          cnt  = (float*)(ws + o_cnt);
  unsigned short* molX = (unsigned short*)(ws + o_molX);
  unsigned short* MH   = (unsigned short*)(ws + o_MH);

  const int TB = 256;               // 8 waves per block
  const int WPB = TB / 32;

  // ---- 1) weight conversions to bf16 col-major padded ----
  int d_in1 = FDIM + HDIM;          // 279
  convert_wcm_kernel<<<cdiv((size_t)FFN_H * K1PAD, TB), TB, 0, stream>>>(W1_aa, w1aa, d_in1, FFN_H, K1PAD, FFN_H);
  convert_wcm_kernel<<<cdiv((size_t)HDIM  * K2PAD, TB), TB, 0, stream>>>(W2_aa, w2aa, FFN_H, HDIM, K2PAD, HDIM);
  convert_wcm_kernel<<<cdiv((size_t)FFN_H * K1PAD, TB), TB, 0, stream>>>(W1_ab, w1ab, d_in1, FFN_H, K1PAD, FFN_H);
  convert_wcm_kernel<<<cdiv((size_t)HDIM  * K2PAD, TB), TB, 0, stream>>>(W2_ab, w2ab, FFN_H, HDIM, K2PAD, HDIM);
  convert_wcm_kernel<<<cdiv((size_t)FFN_H * KFPAD, TB), TB, 0, stream>>>(Wf1, wf1, HDIM + FEAT, FFN_H, KFPAD, FFN_H);
  convert_wcm_kernel<<<cdiv((size_t)NFPAD * K2PAD, TB), TB, 0, stream>>>(Wf2, wf2, FFN_H, TASKS, K2PAD, NFPAD);

  unsigned int atomBlocks = cdiv(N_ATOMS, WPB);
  // waves = (M/16) * (Npad/16/NT), NT=4 for the wide GEMMs
  unsigned int g1Blocks = cdiv((size_t)(N_ATOMS / 16) * (FFN_H / 64), WPB);
  unsigned int g2Blocks = cdiv((size_t)(N_ATOMS / 16) * (HDIM / 64), WPB);

  // ---- 2) branch A: atom-from-atom -> d_out atom block ----
  prep_atoms_kernel<<<atomBlocks, TB, 0, stream>>>(atom_output, orig, a2a, Xbuf);
  wmma_gemm_kernel<4, K1PAD><<<g1Blocks, TB, 0, stream>>>(Xbuf, w1aa, b1_aa, H1, N_ATOMS, FFN_H, FFN_H, 0);
  wmma_gemm_kernel<4, K2PAD><<<g2Blocks, TB, 0, stream>>>(H1, w2aa, b2_aa, F2, N_ATOMS, HDIM, HDIM, 1);
  layernorm128_kernel<<<atomBlocks, TB, 0, stream>>>(F2, gamma_a, beta_a, out_atom, N_ATOMS);

  // ---- 3) branch B: atom-from-bond (buffers reused) ----
  prep_atoms_kernel<<<atomBlocks, TB, 0, stream>>>(bond_output, orig, a2b, Xbuf);
  wmma_gemm_kernel<4, K1PAD><<<g1Blocks, TB, 0, stream>>>(Xbuf, w1ab, b1_ab, H1, N_ATOMS, FFN_H, FFN_H, 0);
  wmma_gemm_kernel<4, K2PAD><<<g2Blocks, TB, 0, stream>>>(H1, w2ab, b2_ab, F2, N_ATOMS, HDIM, HDIM, 1);
  layernorm128_kernel<<<atomBlocks, TB, 0, stream>>>(F2, gamma_b, beta_b, F2, N_ATOMS);  // in-place

  // ---- 4) segment mean readout ----
  size_t nz = (size_t)10000 * HDIM + 10000;       // msum and cnt are contiguous
  zero_kernel<<<cdiv(nz, TB), TB, 0, stream>>>(msum, nz);
  seg_reduce_kernel<<<atomBlocks, TB, 0, stream>>>(F2, seg, msum, cnt);
  mol_prep_kernel<<<cdiv(10000, WPB), TB, 0, stream>>>(msum, cnt, feats, molX, 10000);

  // ---- 5) mol FFN + sigmoid -> d_out mol block ----
  unsigned int m1Blocks = cdiv((size_t)(10000 / 16) * (FFN_H / 64), WPB);
  unsigned int m2Blocks = cdiv((size_t)(10000 / 16) * (NFPAD / 16), WPB);
  wmma_gemm_kernel<4, KFPAD><<<m1Blocks, TB, 0, stream>>>(molX, wf1, bf1, MH, 10000, FFN_H, FFN_H, 0);
  wmma_gemm_kernel<1, K2PAD><<<m2Blocks, TB, 0, stream>>>(MH, wf2, bf2, out_mol, 10000, NFPAD, TASKS, 2);
}
